// AqtDotGeneral_51634096832568
// MI455X (gfx1250) — 8-bit matrix operand order repaired and checked by reading, compiled, not run on hardware
//
#include <hip/hip_runtime.h>
#include <stdint.h>

#define MDIM 8192
#define NDIM 4096
#define KDIM 4096

typedef __attribute__((ext_vector_type(8))) int v8i;

__global__ __launch_bounds__(256) void quant_lhs_kernel(
    const float* __restrict__ lhs, int8_t* __restrict__ lhs_q,
    float* __restrict__ lhs_s) {
  const int row = blockIdx.x;
  const int t = threadIdx.x;
  const float* rp = lhs + (size_t)row * KDIM;

  float4 v[4];
  float m = 0.0f;
#pragma unroll
  for (int i = 0; i < 4; ++i) {
    v[i] = ((const float4*)rp)[t * 4 + i];
    m = fmaxf(m, fmaxf(fmaxf(fabsf(v[i].x), fabsf(v[i].y)),
                       fmaxf(fabsf(v[i].z), fabsf(v[i].w))));
  }
#pragma unroll
  for (int off = 16; off > 0; off >>= 1) m = fmaxf(m, __shfl_xor(m, off, 32));

  __shared__ float sred[8];
  __shared__ float sscale;
  const int wave = t >> 5, lane = t & 31;
  if (lane == 0) sred[wave] = m;
  __syncthreads();
  if (t == 0) {
    float mm = sred[0];
#pragma unroll
    for (int i = 1; i < 8; ++i) mm = fmaxf(mm, sred[i]);
    float s = mm * (1.0f / 127.0f);
    if (s == 0.0f) s = 1.0f;
    sscale = s;
    lhs_s[row] = s;
  }
  __syncthreads();
  const float inv = 1.0f / sscale;

  int packed[4];
#pragma unroll
  for (int i = 0; i < 4; ++i) {
    float f[4] = {v[i].x, v[i].y, v[i].z, v[i].w};
    int p = 0;
#pragma unroll
    for (int j = 0; j < 4; ++j) {
      float q = rintf(f[j] * inv);
      q = fminf(fmaxf(q, -127.0f), 127.0f);
      p |= ((int)q & 0xFF) << (8 * j);
    }
    packed[i] = p;
  }
  ((int4*)(lhs_q + (size_t)row * KDIM))[t] =
      make_int4(packed[0], packed[1], packed[2], packed[3]);
}

__global__ __launch_bounds__(256) void rhs_scale_kernel(
    const float* __restrict__ rhs, float* __restrict__ rhs_s) {
  const int n = blockIdx.x * 256 + threadIdx.x;
  float m = 0.0f;
  for (int k = 0; k < KDIM; k += 4) {
    m = fmaxf(m, fabsf(rhs[(size_t)(k + 0) * NDIM + n]));
    m = fmaxf(m, fabsf(rhs[(size_t)(k + 1) * NDIM + n]));
    m = fmaxf(m, fabsf(rhs[(size_t)(k + 2) * NDIM + n]));
    m = fmaxf(m, fabsf(rhs[(size_t)(k + 3) * NDIM + n]));
  }
  float s = m * (1.0f / 127.0f);
  if (s == 0.0f) s = 1.0f;
  rhs_s[n] = s;
}

__global__ __launch_bounds__(256) void quant_rhs_t_kernel(
    const float* __restrict__ rhs, const float* __restrict__ rhs_s,
    int8_t* __restrict__ rhs_qt) {
  __shared__ int8_t tile[64][68];
  const int n0 = blockIdx.x * 64;
  const int k0 = blockIdx.y * 64;
  const int t = threadIdx.x;

  {
    const int kl = t >> 2;
    const int ng = (t & 3) * 16;
    const float* rp = rhs + (size_t)(k0 + kl) * NDIM + n0 + ng;
#pragma unroll
    for (int j = 0; j < 16; ++j) {
      const float s = rhs_s[n0 + ng + j];
      float q = rintf(rp[j] / s);
      q = fminf(fmaxf(q, -127.0f), 127.0f);
      tile[kl][ng + j] = (int8_t)(int)q;
    }
  }
  __syncthreads();
  {
    const int nl = t >> 2;
    const int kg = (t & 3) * 16;
    int packed[4];
#pragma unroll
    for (int w = 0; w < 4; ++w) {
      int p = 0;
#pragma unroll
      for (int j = 0; j < 4; ++j)
        p |= ((int)(uint8_t)tile[kg + w * 4 + j][nl]) << (8 * j);
      packed[w] = p;
    }
    ((int4*)(rhs_qt + (size_t)(n0 + nl) * KDIM + k0 + kg))[0] =
        make_int4(packed[0], packed[1], packed[2], packed[3]);
  }
}

__global__ __launch_bounds__(256) void gemm_iu8_kernel(
    const int8_t* __restrict__ Aq, const int8_t* __restrict__ Bqt,
    const float* __restrict__ As, const float* __restrict__ Bs,
    float* __restrict__ out) {
  const int t = threadIdx.x;
  const int wave = t >> 5;
  const int lane = t & 31;
  const int lmod = lane & 15;
  const int lhi = lane >> 4;

  const int bm = blockIdx.y * 128;
  const int bn = blockIdx.x * 128;
  const int m0 = bm + (wave & 3) * 32;
  const int n0 = bn + (wave >> 2) * 64;

  v8i acc[2][4];
#pragma unroll
  for (int i = 0; i < 2; ++i)
#pragma unroll
    for (int j = 0; j < 4; ++j) acc[i][j] = (v8i)0;

  const size_t arow[2] = {(size_t)(m0 + lmod) * KDIM,
                          (size_t)(m0 + 16 + lmod) * KDIM};

  for (int k = 0; k < KDIM; k += 64) {
    v8i a[2];
    const int ak = k + lhi * 16;
#pragma unroll
    for (int fm = 0; fm < 2; ++fm) {
      const int8_t* p = Aq + arow[fm] + ak;
#pragma unroll
      for (int j = 0; j < 4; ++j) {
        const int2 d = *(const int2*)(p + (j & 1) * 8 + (j >> 1) * 32);
        a[fm][2 * j + 0] = d.x;
        a[fm][2 * j + 1] = d.y;
      }
    }
    const int bk = k + lhi * 16;
#pragma unroll
    for (int fn = 0; fn < 4; ++fn) {
      const int8_t* p = Bqt + (size_t)(n0 + fn * 16 + lmod) * KDIM + bk;
      const int4 lo = *(const int4*)(p);
      const int4 hi = *(const int4*)(p + 32);
      v8i b;
      b[0] = lo.x; b[1] = lo.y; b[2] = lo.z; b[3] = lo.w;
      b[4] = hi.x; b[5] = hi.y; b[6] = hi.z; b[7] = hi.w;
#pragma unroll
      for (int fm = 0; fm < 2; ++fm)
        acc[fm][fn] = __builtin_amdgcn_wmma_i32_16x16x64_iu8(
             true, a[fm],  true, b, acc[fm][fn],
             false,  false);
    }
  }

#pragma unroll
  for (int fn = 0; fn < 4; ++fn) {
    const int c = n0 + fn * 16 + lmod;
    const float rs = Bs[c];
#pragma unroll
    for (int fm = 0; fm < 2; ++fm) {
      const int rbase = m0 + fm * 16 + lhi * 8;
#pragma unroll
      for (int v = 0; v < 8; ++v) {
        const int r = rbase + v;
        out[(size_t)r * NDIM + c] = (float)acc[fm][fn][v] * As[r] * rs;
      }
    }
  }
}

extern "C" void kernel_launch(void* const* d_in, const int* in_sizes, int n_in,
                              void* d_out, int out_size, void* d_ws,
                              size_t ws_size, hipStream_t stream) {
  const float* lhs = (const float*)d_in[0];
  const float* rhs = (const float*)d_in[1];
  float* out = (float*)d_out;

  int8_t* ws = (int8_t*)d_ws;
  int8_t* lhs_q = ws;
  int8_t* rhs_qt = ws + (size_t)MDIM * KDIM;
  float* lhs_s = (float*)(ws + (size_t)MDIM * KDIM + (size_t)NDIM * KDIM);
  float* rhs_s = lhs_s + MDIM;

  quant_lhs_kernel<<<MDIM, 256, 0, stream>>>(lhs, lhs_q, lhs_s);
  rhs_scale_kernel<<<NDIM / 256, 256, 0, stream>>>(rhs, rhs_s);
  quant_rhs_t_kernel<<<dim3(NDIM / 64, KDIM / 64), 256, 0, stream>>>(
      rhs, rhs_s, rhs_qt);
  gemm_iu8_kernel<<<dim3(NDIM / 128, MDIM / 128), 256, 0, stream>>>(
      lhs_q, rhs_qt, lhs_s, rhs_s, out);
}
